// MultiHeadAttentionV2_82875688944191
// MI455X (gfx1250) — matrix operand order repaired and checked by reading, not compiled, not run on hardware
//
#include <hip/hip_runtime.h>
#include <hip/hip_bf16.h>
#include <stdint.h>

// Problem constants: x[4,2048,1024], w_attn[1024,3072], w_proj[1024,1024]
#define B_  4
#define T_  2048
#define C_  1024
#define H_  16
#define D_  64

typedef __attribute__((ext_vector_type(16))) __bf16 v16bf;
typedef __attribute__((ext_vector_type(8)))  __bf16 v8bf;
typedef __attribute__((ext_vector_type(8)))  float  v8f;

// fp32 -> bf16 round-to-nearest-even (scalar fallback)
__device__ __forceinline__ unsigned short f2bf(float f) {
  union { float f; uint32_t u; } x; x.f = f;
  uint32_t u = x.u;
  uint32_t r = u + 0x7FFFu + ((u >> 16) & 1u);
  return (unsigned short)(r >> 16);
}

// Packed pair conversion: prefer single-instruction v_cvt_pk_bf16_f32.
__device__ __forceinline__ uint32_t pk_bf16(float a, float b) {
#if __has_builtin(__builtin_amdgcn_cvt_pk_bf16_f32)
  auto r = __builtin_amdgcn_cvt_pk_bf16_f32(a, b);
  uint32_t u;
  __builtin_memcpy(&u, &r, 4);
  return u;
#else
  return (uint32_t)f2bf(a) | ((uint32_t)f2bf(b) << 16);
#endif
}

// Build a 16-element bf16 fragment from two 16-byte-aligned 8-element chunks.
__device__ __forceinline__ v16bf ldfrag(const unsigned short* p0, const unsigned short* p1) {
  v8bf lo = *(const v8bf*)p0;
  v8bf hi = *(const v8bf*)p1;
  return __builtin_shufflevector(lo, hi, 0,1,2,3,4,5,6,7,8,9,10,11,12,13,14,15);
}

__device__ __forceinline__ v8f wmma_bf16(v16bf a, v16bf b, v8f c) {
  return __builtin_amdgcn_wmma_f32_16x16x32_bf16(false, a, false, b, (short)0, c, false, false);
}

// --- CDNA5 async copy global->LDS (ASYNCcnt-tracked, CDNA5 ISA) -----------
__device__ __forceinline__ void async_b128(uint32_t lds_byte_addr, uint64_t gaddr) {
  asm volatile("global_load_async_to_lds_b128 %0, %1, off"
               :: "v"(lds_byte_addr), "v"(gaddr) : "memory");
}
__device__ __forceinline__ void wait_async0() {
  asm volatile("s_wait_asynccnt 0" ::: "memory");
}

// Stage one 128x32 bf16 tile (row pitch `pitch` elems in global, LDS pitch 40
// elems = 80B) using 2 async B128 copies per thread (256 threads).
__device__ __forceinline__ void stage_tile_async(const unsigned short* gbase, int pitch,
                                                 uint32_t lds_base, int tid) {
#pragma unroll
  for (int i = 0; i < 2; i++) {
    int c   = tid + i * 256;       // 0..511 chunks of 16B
    int row = c >> 2;              // 4 chunks per 32-elem row
    int cc  = (c & 3) * 8;         // elem offset in row
    uint64_t g = (uint64_t)(uintptr_t)(gbase + (size_t)row * pitch + cc);
    uint32_t l = lds_base + (uint32_t)(row * 80 + (c & 3) * 16);
    async_b128(l, g);
  }
}

// ---------------------------------------------------------------------------
// Pre-pass 1: streaming fp32 -> bf16 (same layout). One float4 per thread.
// ---------------------------------------------------------------------------
__global__ __launch_bounds__(256) void cvt_f32_bf16(const float* __restrict__ src,
                                                    unsigned short* __restrict__ dst) {
  size_t i = (size_t)blockIdx.x * 256 + threadIdx.x;
  const float4 v = *(const float4*)(src + i * 4);
  uint2 o;
  o.x = pk_bf16(v.x, v.y);
  o.y = pk_bf16(v.z, v.w);
  *(uint2*)(dst + i * 4) = o;
}

// ---------------------------------------------------------------------------
// Pre-pass 2: fp32 [K][N] -> bf16 [N][K] (transpose + convert), one-time.
// Thread: column n, rows k0..k0+3 -> one 8B store.
// ---------------------------------------------------------------------------
__global__ __launch_bounds__(256) void cvt_transpose(const float* __restrict__ src,
                                                     unsigned short* __restrict__ dst,
                                                     int K, int N) {
  int n  = blockIdx.x * 256 + threadIdx.x;
  int k0 = blockIdx.y * 4;
  float a = src[(size_t)(k0 + 0) * N + n];
  float b = src[(size_t)(k0 + 1) * N + n];
  float c = src[(size_t)(k0 + 2) * N + n];
  float d = src[(size_t)(k0 + 3) * N + n];
  uint2 o;
  o.x = pk_bf16(a, b);
  o.y = pk_bf16(c, d);
  *(uint2*)(dst + (size_t)n * K + k0) = o;
}

// ---------------------------------------------------------------------------
// Kernel 1: qkv = x @ w_attn  (M=8192, N=3072, K=1024), all-bf16 operands,
// A [M][K], B pre-transposed [N][K]. Double-buffered async-LDS staging.
// Output scattered: Q,K -> [B,H,T,D]; V -> [B,H,D,T] (transposed for flash).
// ---------------------------------------------------------------------------
__global__ __launch_bounds__(256) void qkv_gemm(const unsigned short* __restrict__ Xb,
                                                const unsigned short* __restrict__ WT,
                                                unsigned short* __restrict__ Qkv) {
  __shared__ unsigned short As[2][128][40];
  __shared__ unsigned short Bs[2][128][40];

  const int tid  = threadIdx.x;
  const int lane = tid & 31;
  const int wid  = tid >> 5;
  const int lr   = lane & 15;
  const int kh   = lane >> 4;
  const int waveM = wid & 3;
  const int waveN = wid >> 2;
  const int m0 = blockIdx.x * 128;
  const int n0 = blockIdx.y * 128;

  const uint32_t aB0 = (uint32_t)(uintptr_t)&As[0][0][0];
  const uint32_t aB1 = (uint32_t)(uintptr_t)&As[1][0][0];
  const uint32_t bB0 = (uint32_t)(uintptr_t)&Bs[0][0][0];
  const uint32_t bB1 = (uint32_t)(uintptr_t)&Bs[1][0][0];

  const unsigned short* Ab = Xb + (size_t)m0 * 1024;
  const unsigned short* Bb = WT + (size_t)n0 * 1024;

  v8f zero = {};
  v8f acc[2][4];
#pragma unroll
  for (int i = 0; i < 2; i++)
#pragma unroll
    for (int j = 0; j < 4; j++) acc[i][j] = zero;

  stage_tile_async(Ab, 1024, aB0, tid);
  stage_tile_async(Bb, 1024, bB0, tid);

  for (int kt = 0; kt < 32; kt++) {
    wait_async0();
    __syncthreads();
    const int cur = kt & 1;
    if (kt + 1 < 32) {
      stage_tile_async(Ab + (kt + 1) * 32, 1024, cur ? aB0 : aB1, tid);
      stage_tile_async(Bb + (kt + 1) * 32, 1024, cur ? bB0 : bB1, tid);
    }
    if (kt + 2 < 32) {  // pull tile k+2 toward L2 while async fills k+1
      __builtin_prefetch(Ab + (size_t)(tid >> 1) * 1024 + (kt + 2) * 32, 0, 1);
      __builtin_prefetch(Bb + (size_t)(tid >> 1) * 1024 + (kt + 2) * 32, 0, 1);
    }

    v16bf a[2];
#pragma unroll
    for (int i = 0; i < 2; i++) {
      const unsigned short* p = &As[cur][waveM * 32 + i * 16 + lr][kh * 8];
      a[i] = ldfrag(p, p + 16);
    }
    v16bf b[4];
#pragma unroll
    for (int j = 0; j < 4; j++) {
      const unsigned short* p = &Bs[cur][waveN * 64 + j * 16 + lr][kh * 8];
      b[j] = ldfrag(p, p + 16);
    }
#pragma unroll
    for (int i = 0; i < 2; i++)
#pragma unroll
      for (int j = 0; j < 4; j++)
        acc[i][j] = wmma_bf16(a[i], b[j], acc[i][j]);
  }

  const size_t seg = (size_t)B_ * H_ * T_ * D_;
#pragma unroll
  for (int i = 0; i < 2; i++) {
#pragma unroll
    for (int j = 0; j < 4; j++) {
#pragma unroll
      for (int r = 0; r < 8; r++) {
        int m = m0 + waveM * 32 + i * 16 + r + 8 * kh;   // C layout row
        int n = n0 + waveN * 64 + j * 16 + lr;           // C layout col
        int b_ = m >> 11, t = m & 2047;
        int sel = n >> 10, c0 = n & 1023;
        int h = c0 >> 6, d = c0 & 63;
        size_t dst;
        if (sel == 2)  // V transposed: [B,H,D,T]
          dst = 2 * seg + ((((size_t)b_ * H_ + h) * D_ + d) * T_ + t);
        else
          dst = (size_t)sel * seg + ((((size_t)b_ * H_ + h) * T_ + t) * D_ + d);
        Qkv[dst] = f2bf(acc[i][j][r]);
      }
    }
  }
}

// ---------------------------------------------------------------------------
// Kernel 2: causal flash attention per (b,h). 4 waves x 16 q-rows.
// Q,K fragments direct from global [B,H,T,D]; V direct from global [B,H,D,T]
// (pre-transposed by qkv_gemm) -> no LDS staging, no block barriers.
// P relayout C->A via per-wave LDS (same-wave DS ordering).
// ---------------------------------------------------------------------------
__global__ __launch_bounds__(128) void flash_attn(const unsigned short* __restrict__ Qw,
                                                  const unsigned short* __restrict__ Kw,
                                                  const unsigned short* __restrict__ Vw,
                                                  unsigned short* __restrict__ Yw) {
  __shared__ unsigned short Pl[4][16][40];   // per-wave P tile [row][key]

  const int tid  = threadIdx.x;
  const int lane = tid & 31;
  const int w    = tid >> 5;
  const int lr   = lane & 15;
  const int kh   = lane >> 4;

  const int bh  = blockIdx.y;               // b*H + h
  const int qb  = blockIdx.x * 64;
  const int qr0 = qb + w * 16;

  const unsigned short* Qh  = Qw + (size_t)bh * T_ * D_;
  const unsigned short* Kh  = Kw + (size_t)bh * T_ * D_;
  const unsigned short* VTh = Vw + (size_t)bh * D_ * T_;   // [D][T]

  v16bf aQ[2];
  {
    const unsigned short* qp = Qh + (size_t)(qr0 + lr) * D_;
#pragma unroll
    for (int s = 0; s < 2; s++)
      aQ[s] = ldfrag(qp + 32 * s + 8 * kh, qp + 32 * s + 16 + 8 * kh);
  }

  const float sc = 0.125f * 1.44269504f;    // 1/sqrt(64) * log2(e)

  float mrow[8], lrow[8];
#pragma unroll
  for (int r = 0; r < 8; r++) { mrow[r] = -3.0e38f; lrow[r] = 0.0f; }
  v8f zero = {};
  v8f oacc[4];
#pragma unroll
  for (int j = 0; j < 4; j++) oacc[j] = zero;

  const int nkt = (qb + 64) >> 5;
  for (int kt = 0; kt < nkt; kt++) {
    const int kb = kt * 32;

    // S = Q K^T, two 16-key column tiles
    v8f sC[2];
#pragma unroll
    for (int t2 = 0; t2 < 2; t2++) {
      sC[t2] = zero;
      const unsigned short* kp = Kh + (size_t)(kb + t2 * 16 + lr) * D_;
#pragma unroll
      for (int s = 0; s < 2; s++) {
        v16bf bK = ldfrag(kp + 32 * s + 8 * kh, kp + 32 * s + 16 + 8 * kh);
        sC[t2] = wmma_bf16(aQ[s], bK, sC[t2]);
      }
    }

    // Online softmax (exp2 domain), causal mask, 16-lane row reductions
    float p[2][8], alpha[8];
#pragma unroll
    for (int r = 0; r < 8; r++) {
      int qrow = qr0 + r + 8 * kh;
      float s0 = sC[0][r] * sc;
      float s1 = sC[1][r] * sc;
      if (kb + lr      > qrow) s0 = -3.0e38f;
      if (kb + 16 + lr > qrow) s1 = -3.0e38f;
      float mx = fmaxf(s0, s1);
#pragma unroll
      for (int off = 1; off < 16; off <<= 1) mx = fmaxf(mx, __shfl_xor(mx, off, 32));
      float mnew = fmaxf(mrow[r], mx);
      alpha[r] = exp2f(mrow[r] - mnew);
      float p0 = exp2f(s0 - mnew);
      float p1 = exp2f(s1 - mnew);
      float ps = p0 + p1;
#pragma unroll
      for (int off = 1; off < 16; off <<= 1) ps += __shfl_xor(ps, off, 32);
      lrow[r] = lrow[r] * alpha[r] + ps;
      mrow[r] = mnew;
      p[0][r] = p0; p[1][r] = p1;
    }

    // P: C-layout -> A-layout via per-wave LDS
#pragma unroll
    for (int t2 = 0; t2 < 2; t2++)
#pragma unroll
      for (int r = 0; r < 8; r++)
        Pl[w][r + 8 * kh][t2 * 16 + lr] = f2bf(p[t2][r]);

    const unsigned short* pp = &Pl[w][lr][kh * 8];
    v16bf aP = ldfrag(pp, pp + 16);

    // O = O*alpha + P @ V ; V^T rows are contiguous in keys
#pragma unroll
    for (int j = 0; j < 4; j++) {
#pragma unroll
      for (int r = 0; r < 8; r++) oacc[j][r] *= alpha[r];
      const unsigned short* vp = VTh + (size_t)(j * 16 + lr) * T_ + kb + 8 * kh;
      v16bf bV = ldfrag(vp, vp + 16);
      oacc[j] = wmma_bf16(aP, bV, oacc[j]);
    }
  }

  // Normalize (one reciprocal per row) and write y [B,T,C] bf16
  const int b_ = bh >> 4, h = bh & 15;
#pragma unroll
  for (int r = 0; r < 8; r++) {
    int t = qr0 + r + 8 * kh;
    float inv = 1.0f / lrow[r];
    size_t rowbase = ((size_t)b_ * T_ + t) * C_ + h * D_ + lr;
#pragma unroll
    for (int j = 0; j < 4; j++)
      Yw[rowbase + j * 16] = f2bf(oacc[j][r] * inv);
  }
}

// ---------------------------------------------------------------------------
// Kernel 3: out = y @ w_proj (M=8192, N=1024, K=1024), bf16 A [M][K],
// bf16 B pre-transposed [N][K], fp32 out. Same async double-buffered engine.
// ---------------------------------------------------------------------------
__global__ __launch_bounds__(256) void proj_gemm(const unsigned short* __restrict__ Yw,
                                                 const unsigned short* __restrict__ WT,
                                                 float* __restrict__ Out) {
  __shared__ unsigned short As[2][128][40];
  __shared__ unsigned short Bs[2][128][40];

  const int tid  = threadIdx.x;
  const int lane = tid & 31;
  const int wid  = tid >> 5;
  const int lr   = lane & 15;
  const int kh   = lane >> 4;
  const int waveM = wid & 3;
  const int waveN = wid >> 2;
  const int m0 = blockIdx.x * 128;
  const int n0 = blockIdx.y * 128;

  const uint32_t aB0 = (uint32_t)(uintptr_t)&As[0][0][0];
  const uint32_t aB1 = (uint32_t)(uintptr_t)&As[1][0][0];
  const uint32_t bB0 = (uint32_t)(uintptr_t)&Bs[0][0][0];
  const uint32_t bB1 = (uint32_t)(uintptr_t)&Bs[1][0][0];

  const unsigned short* Ab = Yw + (size_t)m0 * 1024;
  const unsigned short* Bb = WT + (size_t)n0 * 1024;

  v8f zero = {};
  v8f acc[2][4];
#pragma unroll
  for (int i = 0; i < 2; i++)
#pragma unroll
    for (int j = 0; j < 4; j++) acc[i][j] = zero;

  stage_tile_async(Ab, 1024, aB0, tid);
  stage_tile_async(Bb, 1024, bB0, tid);

  for (int kt = 0; kt < 32; kt++) {
    wait_async0();
    __syncthreads();
    const int cur = kt & 1;
    if (kt + 1 < 32) {
      stage_tile_async(Ab + (kt + 1) * 32, 1024, cur ? aB0 : aB1, tid);
      stage_tile_async(Bb + (kt + 1) * 32, 1024, cur ? bB0 : bB1, tid);
    }
    if (kt + 2 < 32) {
      __builtin_prefetch(Ab + (size_t)(tid >> 1) * 1024 + (kt + 2) * 32, 0, 1);
      __builtin_prefetch(Bb + (size_t)(tid >> 1) * 1024 + (kt + 2) * 32, 0, 1);
    }

    v16bf a[2];
#pragma unroll
    for (int i = 0; i < 2; i++) {
      const unsigned short* p = &As[cur][waveM * 32 + i * 16 + lr][kh * 8];
      a[i] = ldfrag(p, p + 16);
    }
    v16bf b[4];
#pragma unroll
    for (int j = 0; j < 4; j++) {
      const unsigned short* p = &Bs[cur][waveN * 64 + j * 16 + lr][kh * 8];
      b[j] = ldfrag(p, p + 16);
    }
#pragma unroll
    for (int i = 0; i < 2; i++)
#pragma unroll
      for (int j = 0; j < 4; j++)
        acc[i][j] = wmma_bf16(a[i], b[j], acc[i][j]);
  }

#pragma unroll
  for (int i = 0; i < 2; i++) {
#pragma unroll
    for (int j = 0; j < 4; j++) {
#pragma unroll
      for (int r = 0; r < 8; r++) {
        int m = m0 + waveM * 32 + i * 16 + r + 8 * kh;
        int n = n0 + waveN * 64 + j * 16 + lr;
        Out[(size_t)m * 1024 + n] = acc[i][j][r];
      }
    }
  }
}

// ---------------------------------------------------------------------------
extern "C" void kernel_launch(void* const* d_in, const int* in_sizes, int n_in,
                              void* d_out, int out_size, void* d_ws, size_t ws_size,
                              hipStream_t stream) {
  (void)in_sizes; (void)n_in; (void)out_size; (void)ws_size;
  const float* x      = (const float*)d_in[0];
  const float* w_attn = (const float*)d_in[1];
  const float* w_proj = (const float*)d_in[2];
  float* out = (float*)d_out;

  const size_t seg = (size_t)B_ * H_ * T_ * D_;      // 8,388,608 elems
  unsigned short* Qw  = (unsigned short*)d_ws;
  unsigned short* Kw  = Qw + seg;
  unsigned short* Vw  = Kw + seg;                    // [B,H,D,T]
  unsigned short* Yw  = Vw + seg;                    // [B,T,C]
  unsigned short* Xb  = Yw + seg;                    // x in bf16 [8192][1024]
  unsigned short* WaT = Xb + seg;                    // w_attn^T bf16 [3072][1024]
  unsigned short* WpT = WaT + (size_t)3072 * 1024;   // w_proj^T bf16 [1024][1024]

  // One-time convert / transpose pre-passes (bandwidth-trivial)
  cvt_f32_bf16<<<8192, 256, 0, stream>>>(x, Xb);
  cvt_transpose<<<dim3(12, 256), 256, 0, stream>>>(w_attn, WaT, 1024, 3072);
  cvt_transpose<<<dim3(4, 256), 256, 0, stream>>>(w_proj, WpT, 1024, 1024);

  // qkv = x @ w_attn -> Q,K [B,H,T,D]; V [B,H,D,T]
  qkv_gemm<<<dim3(64, 24), 256, 0, stream>>>(Xb, WaT, Qw);
  // causal flash attention -> Yw [B,T,C]
  flash_attn<<<dim3(32, 64), 128, 0, stream>>>(Qw, Kw, Vw, Yw);
  // out = Yw @ w_proj
  proj_gemm<<<dim3(64, 8), 256, 0, stream>>>(Yw, WpT, out);
}
